// MultiHeadMixAttention_49529562858149
// MI455X (gfx1250) — matrix operand order repaired and checked by reading, not compiled, not run on hardware
//
#include <hip/hip_runtime.h>
#include <hip/hip_bf16.h>
#include <stdint.h>

// MultiHeadMixAttention for MI455X (gfx1250, wave32, WMMA bf16 + TDM path).
// Pipeline: fp32->bf16 casts, 4 WMMA projection GEMMs (TDM double-buffered
// LDS staging), 2 symmetric flash attention passes (row softmax / column
// softmax via operand swap, K tile staged by the Tensor Data Mover),
// 2 WMMA output-projection GEMMs (fp32 out).

typedef __attribute__((ext_vector_type(16))) __bf16 v16bf;
typedef __attribute__((ext_vector_type(8)))  float  v8f;
typedef unsigned int u32x4 __attribute__((ext_vector_type(4)));
typedef int          i32x4 __attribute__((ext_vector_type(4)));
typedef int          i32x8 __attribute__((ext_vector_type(8)));

#if defined(__has_builtin)
#if __has_builtin(__builtin_amdgcn_tensor_load_to_lds) && \
    __has_builtin(__builtin_amdgcn_s_wait_tensorcnt)
#define USE_TDM 1
#endif
#endif
#ifndef USE_TDM
#define USE_TDM 0
#endif

#define H_ 16
#define D_ 64
#define E_ 1024
#define LOG2E 1.44269504088896340736f

#define WMMA_BF16(a, b, c) \
  __builtin_amdgcn_wmma_f32_16x16x32_bf16(false, (a), false, (b), (short)0, (c), false, false)

struct FragBF { union { v16bf v; uint32_t u[8]; }; };

// CDNA5 16-bit A-matrix 16x32 layout: lanes 0-15 rows M=0..15, K-octet by lane-half.
__device__ __forceinline__ int a_k_of_j(int j, int lane) {
  int khalf = (lane >> 4) << 3;                 // 0 or 8
  return (j < 4) ? (khalf + 2 * j) : (16 + khalf + 2 * (j - 4));
}
// CDNA5 16-bit B-matrix 32x16 layout: N = lane&15, K-octet by lane-half (same as A).
__device__ __forceinline__ int b_k_of_j(int j, int lane) {
  int khalf = (lane >> 4) << 3;                 // 0 or 8
  return (j < 4) ? (khalf + 2 * j) : (16 + khalf + 2 * (j - 4));
}

#if USE_TDM
// Low 32 bits of a flat LDS pointer = LDS byte offset (ISA: LDS aperture keeps
// the offset in ADDR[31:0]).
__device__ __forceinline__ unsigned lds_addr_of(const void* p) {
  return (unsigned)(unsigned long long)p;
}

// Issue a TENSOR_LOAD_TO_LDS of a 2-D bf16 tile [tile_h rows x tile_w elems]
// from global (row stride row_stride elems) into LDS at lds_addr, with
// optional descriptor-side padding (pad every 2^(pad_interval+1) DWORDs by
// (pad_amount+1) DWORDs) so the LDS row stride matches the compute layout.
// D# bitfields per CDNA5 ISA ch. 8 (groups 0/1; groups 2/3 zero for 2-D).
__device__ __forceinline__ void tdm_load_2d_bf16(
    unsigned lds_addr, const __bf16* gptr,
    unsigned tile_w, unsigned tile_h, unsigned row_stride,
    unsigned pad_interval, unsigned pad_amount, bool pad_en) {
  unsigned long long ga = (unsigned long long)gptr;
  u32x4 g0;
  g0[0] = 1u;                                              // count=1 (valid)
  g0[1] = lds_addr;                                        // lds_addr[31:0]
  g0[2] = (unsigned)ga;                                    // global_addr[31:0]
  g0[3] = (unsigned)((ga >> 32) & 0x01FFFFFFu) | (2u << 30); // [56:32] | type=2
  i32x8 g1;
  g1[0] = (int)((1u << 16) |                               // data_size=1 (2B)
                (pad_en ? (1u << 20) : 0u) |               // pad_enable
                (pad_interval << 22) | (pad_amount << 25));
  g1[1] = (int)((tile_w & 0xFFFFu) << 16);                 // tensor_dim0 lo16
  g1[2] = (int)((tile_w >> 16) | ((tile_h & 0xFFFFu) << 16)); // dim0 hi | dim1 lo
  g1[3] = (int)((tile_h >> 16) | ((tile_w & 0xFFFFu) << 16)); // dim1 hi | tile_dim0
  g1[4] = (int)(tile_h & 0xFFFFu);                         // tile_dim1 (dim2=0)
  g1[5] = (int)row_stride;                                 // tensor_dim0_stride lo
  g1[6] = 0;                                               // stride hi | dim1_stride
  g1[7] = 0;
  i32x4 gz = {0, 0, 0, 0};
#if __clang_major__ >= 23
  i32x8 gz8 = {0, 0, 0, 0, 0, 0, 0, 0};
  __builtin_amdgcn_tensor_load_to_lds(g0, g1, gz, gz, gz8, 0);
#else
  __builtin_amdgcn_tensor_load_to_lds(g0, g1, gz, gz, 0);
#endif
}
#endif  // USE_TDM

// ---------------------------------------------------------------- casts ----
__global__ __launch_bounds__(256) void cast_bf16_kernel(
    const float* __restrict__ in, __bf16* __restrict__ out, int n) {
  int i = blockIdx.x * 256 + threadIdx.x;
  if (i < n) out[i] = (__bf16)in[i];
}

// W [K=1024, N=1024] fp32 -> Wt [N, K] bf16 (so B-fragments read packed pairs)
__global__ __launch_bounds__(256) void castT_bf16_kernel(
    const float* __restrict__ in, __bf16* __restrict__ out) {
  int id = blockIdx.x * 256 + threadIdx.x;      // 1M threads
  int n = id >> 10, k = id & 1023;
  out[id] = (__bf16)in[k * 1024 + n];
}

// ----------------------------------------------------------------- GEMM ----
// C[M,N] = A[M,K] @ Bt[N,K]^T + bias.  128x128 block tile, 8 waves (4Mx2N),
// each wave 32x64 via 2x4 grid of 16x16x32 bf16 WMMA.  LDS tiles staged by
// the TDM, double buffered: issue tile i+1, s_wait_tensorcnt(2) -> tile i done.
// outF != null  -> fp32 row-major output (final projections)
// outF == null  -> bf16 head-split output [B, H, Lq, D] with scale folded in.
__global__ __launch_bounds__(256) void gemm_bf16_kernel(
    const __bf16* __restrict__ A, const __bf16* __restrict__ Bt,
    const float* __restrict__ bias, float* __restrict__ outF,
    __bf16* __restrict__ outH, int M, int N, int K, float scale, int Lq) {
  __shared__ __attribute__((aligned(16))) __bf16 As[2][128][40];
  __shared__ __attribute__((aligned(16))) __bf16 Bs[2][128][40];
  const int tid = threadIdx.x, lane = tid & 31, wave = tid >> 5;
  const int waveM = wave & 3, waveN = wave >> 2;
  const int m0 = blockIdx.y * 128, n0 = blockIdx.x * 128;

  v8f acc[2][4];
#pragma unroll
  for (int i = 0; i < 2; ++i)
#pragma unroll
    for (int j = 0; j < 4; ++j)
#pragma unroll
      for (int r = 0; r < 8; ++r) acc[i][j][r] = 0.0f;

#if USE_TDM
  // Prime buffer 0.  Tile rows: 32 bf16 = 16 DW -> pad_interval code 3,
  // pad 4 DW (code 3) -> LDS row stride 40 bf16, matching fragment indexing.
  if (wave == 0) {
    tdm_load_2d_bf16(lds_addr_of(&As[0][0][0]), A + (size_t)m0 * K, 32, 128, (unsigned)K, 3, 3, true);
    tdm_load_2d_bf16(lds_addr_of(&Bs[0][0][0]), Bt + (size_t)n0 * K, 32, 128, (unsigned)K, 3, 3, true);
  }
  int ib = 0;
  for (int k0 = 0; k0 < K; k0 += 32, ib ^= 1) {
    if (wave == 0) {
      if (k0 + 32 < K) {
        tdm_load_2d_bf16(lds_addr_of(&As[ib ^ 1][0][0]),
                         A + (size_t)m0 * K + (k0 + 32), 32, 128, (unsigned)K, 3, 3, true);
        tdm_load_2d_bf16(lds_addr_of(&Bs[ib ^ 1][0][0]),
                         Bt + (size_t)n0 * K + (k0 + 32), 32, 128, (unsigned)K, 3, 3, true);
        __builtin_amdgcn_s_wait_tensorcnt(2);   // current pair complete
      } else {
        __builtin_amdgcn_s_wait_tensorcnt(0);
      }
    }
    __syncthreads();
#else
  const int lr = tid >> 1, lc = (tid & 1) << 4;
  const int ib = 0;
  for (int k0 = 0; k0 < K; k0 += 32) {
    const __bf16* ga = A + (size_t)(m0 + lr) * K + k0 + lc;
    const __bf16* gb = Bt + (size_t)(n0 + lr) * K + k0 + lc;
    *(uint4*)&As[0][lr][lc]     = *(const uint4*)ga;
    *(uint4*)&As[0][lr][lc + 8] = *(const uint4*)(ga + 8);
    *(uint4*)&Bs[0][lr][lc]     = *(const uint4*)gb;
    *(uint4*)&Bs[0][lr][lc + 8] = *(const uint4*)(gb + 8);
    __syncthreads();
#endif

    FragBF af[2], bfr[4];
#pragma unroll
    for (int mt = 0; mt < 2; ++mt) {
      int row = waveM * 32 + mt * 16 + (lane & 15);
#pragma unroll
      for (int j = 0; j < 8; ++j)
        af[mt].u[j] = *(const uint32_t*)&As[ib][row][a_k_of_j(j, lane)];
    }
#pragma unroll
    for (int nt = 0; nt < 4; ++nt) {
      int n = waveN * 64 + nt * 16 + (lane & 15);
#pragma unroll
      for (int j = 0; j < 8; ++j)
        bfr[nt].u[j] = *(const uint32_t*)&Bs[ib][n][b_k_of_j(j, lane)];
    }
#pragma unroll
    for (int mt = 0; mt < 2; ++mt)
#pragma unroll
      for (int nt = 0; nt < 4; ++nt)
        acc[mt][nt] = WMMA_BF16(af[mt].v, bfr[nt].v, acc[mt][nt]);
    __syncthreads();
  }

#pragma unroll
  for (int mt = 0; mt < 2; ++mt)
#pragma unroll
    for (int nt = 0; nt < 4; ++nt)
#pragma unroll
      for (int r = 0; r < 8; ++r) {
        int m = m0 + waveM * 32 + mt * 16 + r + ((lane >> 4) << 3);
        int n = n0 + waveN * 64 + nt * 16 + (lane & 15);
        float v = (acc[mt][nt][r] + bias[n]) * scale;
        if (outF) {
          outF[(size_t)m * N + n] = v;
        } else {
          int b = m / Lq, l = m % Lq, h = n >> 6, d = n & 63;
          outH[(((size_t)b * H_ + h) * Lq + l) * D_ + d] = (__bf16)v;
        }
      }
}

// ---------------------------------------------------------------- flash ----
// Per block: 128 query rows of one (b,h). 8 waves x 16 rows. Online softmax
// over Lk; output ctx written bf16 into [B, Lq, E] at column h*64.
#if USE_TDM
#define KSTRIDE 66   // TDM pads each 128B row by 4B -> 66 bf16 row stride
#else
#define KSTRIDE 72   // cooperative path keeps 16B-aligned rows for uint4
#endif
#define PSTRIDE 136

__global__ __launch_bounds__(256) void flash_kernel(
    const __bf16* __restrict__ Q, const __bf16* __restrict__ Km,
    const __bf16* __restrict__ Vm, __bf16* __restrict__ Octx,
    int Lq, int Lk) {
  // Ks (128 x KSTRIDE) overlaid with Ps (128 x 136); Vt holds V transposed.
  __shared__ __attribute__((aligned(16))) unsigned char smemA[128 * PSTRIDE * 2];
  __shared__ __attribute__((aligned(16))) __bf16 Vt[64][PSTRIDE];
  __bf16(*Ks)[KSTRIDE] = (__bf16(*)[KSTRIDE])smemA;
  __bf16(*Ps)[PSTRIDE] = (__bf16(*)[PSTRIDE])smemA;

  const int tid = threadIdx.x, lane = tid & 31, wave = tid >> 5;
  const int bh = blockIdx.y, b = bh >> 4, h = bh & 15;
  const int l0 = blockIdx.x * 128;
  const int rowBase = wave * 16;

  // Q fragments held in registers for the whole Lk loop (D=64 -> 2 K-steps).
  FragBF qf[2];
  const __bf16* qrow = Q + ((size_t)bh * Lq + l0 + rowBase + (lane & 15)) * D_;
#pragma unroll
  for (int kt = 0; kt < 2; ++kt)
#pragma unroll
    for (int j = 0; j < 8; ++j)
      qf[kt].u[j] = *(const uint32_t*)(qrow + kt * 32 + a_k_of_j(j, lane));

  v8f ctx[4];
#pragma unroll
  for (int i = 0; i < 4; ++i)
#pragma unroll
    for (int r = 0; r < 8; ++r) ctx[i][r] = 0.0f;
  float mstat[8], lstat[8];
#pragma unroll
  for (int r = 0; r < 8; ++r) { mstat[r] = -1e30f; lstat[r] = 0.0f; }

  const int lr = tid >> 1, lc = (tid & 1) * 32;

  for (int s0 = 0; s0 < Lk; s0 += 128) {
    // ---- stage K tile (TDM if available) and V tile transposed ----
    const __bf16* gkbase = Km + ((size_t)bh * Lk + s0) * D_;
#if USE_TDM
    if (wave == 0) {
      // 64 bf16 = 32 DW per row -> pad_interval code 4, pad 1 DW -> stride 66.
      tdm_load_2d_bf16(lds_addr_of(&Ks[0][0]), gkbase, 64, 128, 64, 4, 0, true);
    }
#else
    {
      const __bf16* gk = gkbase + (size_t)lr * D_ + lc;
#pragma unroll
      for (int q4 = 0; q4 < 4; ++q4)
        *(uint4*)&Ks[lr][lc + q4 * 8] = *(const uint4*)(gk + q4 * 8);
    }
#endif
    {
      const __bf16* gv = Vm + ((size_t)bh * Lk + s0 + lr) * D_ + lc;
      __bf16 tmp[32];
#pragma unroll
      for (int q4 = 0; q4 < 4; ++q4)
        *(uint4*)&tmp[q4 * 8] = *(const uint4*)(gv + q4 * 8);
#pragma unroll
      for (int i = 0; i < 32; ++i) Vt[lc + i][lr] = tmp[i];
    }
#if USE_TDM
    if (wave == 0) __builtin_amdgcn_s_wait_tensorcnt(0);
#endif
    __syncthreads();

    // ---- scores: wave's 16 rows x full 128 key columns ----
    v8f S[8];
#pragma unroll
    for (int nt = 0; nt < 8; ++nt)
#pragma unroll
      for (int r = 0; r < 8; ++r) S[nt][r] = 0.0f;
#pragma unroll
    for (int kt = 0; kt < 2; ++kt) {
#pragma unroll
      for (int nt = 0; nt < 8; ++nt) {
        FragBF bfr;
        int n = nt * 16 + (lane & 15);
#pragma unroll
        for (int j = 0; j < 8; ++j)
          bfr.u[j] = *(const uint32_t*)&Ks[n][kt * 32 + b_k_of_j(j, lane)];
        S[nt] = WMMA_BF16(qf[kt].v, bfr.v, S[nt]);
      }
    }
    __syncthreads();  // done reading Ks; region reused as Ps

    // ---- online softmax (stats per vgpr slot / lane-half; xor<=8 stays
    //      within a 16-lane half, halves hold disjoint rows) ----
    float rmax[8];
#pragma unroll
    for (int r = 0; r < 8; ++r) {
      float v = S[0][r];
#pragma unroll
      for (int nt = 1; nt < 8; ++nt) v = fmaxf(v, S[nt][r]);
      for (int off = 1; off <= 8; off <<= 1) v = fmaxf(v, __shfl_xor(v, off));
      rmax[r] = v;
    }
    float alpha[8], rsum[8];
#pragma unroll
    for (int r = 0; r < 8; ++r) {
      float mnew = fmaxf(mstat[r], rmax[r]);
      alpha[r] = exp2f((mstat[r] - mnew) * LOG2E);
      mstat[r] = mnew;
      rsum[r] = 0.0f;
    }
#pragma unroll
    for (int nt = 0; nt < 8; ++nt)
#pragma unroll
      for (int r = 0; r < 8; ++r) {
        float p = exp2f((S[nt][r] - mstat[r]) * LOG2E);
        S[nt][r] = p;
        rsum[r] += p;
      }
#pragma unroll
    for (int r = 0; r < 8; ++r) {
      float v = rsum[r];
      for (int off = 1; off <= 8; off <<= 1) v += __shfl_xor(v, off);
      lstat[r] = lstat[r] * alpha[r] + v;
    }
#pragma unroll
    for (int i = 0; i < 4; ++i)
#pragma unroll
      for (int r = 0; r < 8; ++r) ctx[i][r] *= alpha[r];

    // ---- repack P into A-fragment layout via LDS ----
#pragma unroll
    for (int nt = 0; nt < 8; ++nt)
#pragma unroll
      for (int r = 0; r < 8; ++r)
        Ps[rowBase + r + ((lane >> 4) << 3)][nt * 16 + (lane & 15)] = (__bf16)S[nt][r];
    __syncthreads();

    // ---- ctx += P @ V (contraction over 128 keys; V read from Vt so
    //      B-fragments are contiguous packed pairs) ----
#pragma unroll
    for (int ks = 0; ks < 4; ++ks) {
      FragBF pa;
      int prow = rowBase + (lane & 15);
#pragma unroll
      for (int j = 0; j < 8; ++j)
        pa.u[j] = *(const uint32_t*)&Ps[prow][ks * 32 + a_k_of_j(j, lane)];
#pragma unroll
      for (int nt = 0; nt < 4; ++nt) {
        FragBF vb;
        int n = nt * 16 + (lane & 15);
#pragma unroll
        for (int j = 0; j < 8; ++j)
          vb.u[j] = *(const uint32_t*)&Vt[n][ks * 32 + b_k_of_j(j, lane)];
        ctx[nt] = WMMA_BF16(pa.v, vb.v, ctx[nt]);
      }
    }
    __syncthreads();
  }

  // ---- normalize, write ctx to [B, Lq, E] bf16 at columns [h*64, h*64+64) ----
#pragma unroll
  for (int nt = 0; nt < 4; ++nt)
#pragma unroll
    for (int r = 0; r < 8; ++r) {
      int row = l0 + rowBase + r + ((lane >> 4) << 3);
      int d = nt * 16 + (lane & 15);
      float v = ctx[nt][r] / lstat[r];
      Octx[((size_t)b * Lq + row) * E_ + h * D_ + d] = (__bf16)v;
    }
}

// --------------------------------------------------------------- driver ----
extern "C" void kernel_launch(void* const* d_in, const int* in_sizes, int n_in,
                              void* d_out, int out_size, void* d_ws, size_t ws_size,
                              hipStream_t stream) {
  const int B = 2, L = 2048, S = 2048, E = 1024;
  const float* a     = (const float*)d_in[0];
  const float* bIn   = (const float*)d_in[1];
  const float* Wa_qk = (const float*)d_in[2];
  const float* ba_qk = (const float*)d_in[3];
  const float* Wa_v  = (const float*)d_in[4];
  const float* ba_v  = (const float*)d_in[5];
  const float* Wb_qk = (const float*)d_in[6];
  const float* bb_qk = (const float*)d_in[7];
  const float* Wb_v  = (const float*)d_in[8];
  const float* bb_v  = (const float*)d_in[9];
  const float* Wa_o  = (const float*)d_in[10];
  const float* ba_o  = (const float*)d_in[11];
  const float* Wb_o  = (const float*)d_in[12];
  const float* bb_o  = (const float*)d_in[13];
  float* out = (float*)d_out;

  __bf16* ws = (__bf16*)d_ws;
  size_t off = 0;
  const size_t ME = (size_t)B * L * E;   // 4,194,304
  const size_t EE = (size_t)E * E;       // 1,048,576
  __bf16* a_bf  = ws + off; off += ME;
  __bf16* b_bf  = ws + off; off += ME;
  __bf16* WtAqk = ws + off; off += EE;
  __bf16* WtAv  = ws + off; off += EE;
  __bf16* WtBqk = ws + off; off += EE;
  __bf16* WtBv  = ws + off; off += EE;
  __bf16* WtAo  = ws + off; off += EE;
  __bf16* WtBo  = ws + off; off += EE;
  __bf16* aq    = ws + off; off += ME;
  __bf16* av    = ws + off; off += ME;
  __bf16* bq    = ws + off; off += ME;
  __bf16* bv    = ws + off; off += ME;
  __bf16* actx  = ws + off; off += ME;
  __bf16* bctx  = ws + off; off += ME;

  // 1) casts (scale 1/sqrt(D)=0.125 folded into aq projection below)
  cast_bf16_kernel<<<(int)((ME + 255) / 256), 256, 0, stream>>>(a, a_bf, (int)ME);
  cast_bf16_kernel<<<(int)((ME + 255) / 256), 256, 0, stream>>>(bIn, b_bf, (int)ME);
  int tBlocks = (int)(EE / 256);
  castT_bf16_kernel<<<tBlocks, 256, 0, stream>>>(Wa_qk, WtAqk);
  castT_bf16_kernel<<<tBlocks, 256, 0, stream>>>(Wa_v,  WtAv);
  castT_bf16_kernel<<<tBlocks, 256, 0, stream>>>(Wb_qk, WtBqk);
  castT_bf16_kernel<<<tBlocks, 256, 0, stream>>>(Wb_v,  WtBv);
  castT_bf16_kernel<<<tBlocks, 256, 0, stream>>>(Wa_o,  WtAo);
  castT_bf16_kernel<<<tBlocks, 256, 0, stream>>>(Wb_o,  WtBo);

  // 2) projections -> [B,H,N,D] bf16
  dim3 gp(E / 128, (B * L) / 128);  // (8, 32)
  gemm_bf16_kernel<<<gp, 256, 0, stream>>>(a_bf, WtAqk, ba_qk, nullptr, aq, B * L, E, E, 0.125f, L);
  gemm_bf16_kernel<<<gp, 256, 0, stream>>>(a_bf, WtAv,  ba_v,  nullptr, av, B * L, E, E, 1.0f,   L);
  gemm_bf16_kernel<<<gp, 256, 0, stream>>>(b_bf, WtBqk, bb_qk, nullptr, bq, B * S, E, E, 1.0f,   S);
  gemm_bf16_kernel<<<gp, 256, 0, stream>>>(b_bf, WtBv,  bb_v,  nullptr, bv, B * S, E, E, 1.0f,   S);

  // 3) dual flash attention: row softmax then column softmax via operand swap
  dim3 gfa(L / 128, B * H_);  // (16, 32)
  flash_kernel<<<gfa, 256, 0, stream>>>(aq, bq, bv, actx, L, S);
  dim3 gfb(S / 128, B * H_);
  flash_kernel<<<gfb, 256, 0, stream>>>(bq, aq, av, bctx, S, L);

  // 4) output projections -> fp32 d_out (a_out then b_out)
  gemm_bf16_kernel<<<gp, 256, 0, stream>>>(actx, WtAo, ba_o, out, nullptr, B * L, E, E, 1.0f, L);
  gemm_bf16_kernel<<<gp, 256, 0, stream>>>(bctx, WtBo, bb_o, out + ME, nullptr, B * S, E, E, 1.0f, S);
}
